// CrossAttentionWithPosition_3186865733699
// MI455X (gfx1250) — compile-verified, matrix operand order checked by reading, not run on hardware
//
#include <hip/hip_runtime.h>
#include <math.h>
#include <stdint.h>

// ---------------------------------------------------------------------------
// CDNA5 (gfx1250) fused cross-attention-with-position, bf16 WMMA pipeline.
// Wave32; V_WMMA_F32_16X16X32_BF16 for all GEMMs; GLOBAL_LOAD_ASYNC_TO_LDS
// (ASYNCcnt) for tile staging with double buffering.
// ---------------------------------------------------------------------------

typedef __bf16 bf_t;
typedef __attribute__((ext_vector_type(16))) __bf16 v16bf;
typedef __attribute__((ext_vector_type(8)))  float  v8f;
typedef __attribute__((ext_vector_type(4)))  int    v4i;

typedef __attribute__((address_space(1))) v4i as1_v4i;   // global int4
typedef __attribute__((address_space(3))) v4i as3_v4i;   // LDS int4

constexpr int kB    = 32;
constexpr int kNQ   = 1024;
constexpr int kQD   = 1024;
constexpr int kNH   = 16;
constexpr int kHD   = 64;
constexpr int kTEXT = 77;
constexpr int kIMG  = 16;
constexpr int kAUD  = 32;
constexpr int kCTX  = 125;   // TEXT+IMG+AUD
constexpr float kSCALE = 0.125f;  // 64^-0.5

// ---------------------------------------------------------------------------
// Async global->LDS copy of one 16B chunk per lane (wave instruction).
// Builtin when available, inline asm otherwise (same ISA either way).
// ---------------------------------------------------------------------------
#if __has_builtin(__builtin_amdgcn_global_load_async_to_lds_b128) && \
    __has_builtin(__builtin_amdgcn_s_wait_asynccnt)
#define ASYNC_VIA_BUILTIN 1
#else
#define ASYNC_VIA_BUILTIN 0
#endif

__device__ __forceinline__ void async_copy16(const void* g, void* l) {
#if ASYNC_VIA_BUILTIN
  __builtin_amdgcn_global_load_async_to_lds_b128(
      (as1_v4i*)(uintptr_t)g,
      (as3_v4i*)(unsigned int)(uintptr_t)l, 0, 0);
#else
  unsigned int lo = (unsigned int)(uintptr_t)l;           // LDS byte offset
  unsigned long long ga = (unsigned long long)(uintptr_t)g;
  asm volatile("global_load_async_to_lds_b128 %0, %1, off"
               :: "v"(lo), "v"(ga) : "memory");
#endif
}

__device__ __forceinline__ void wait_async0() {
#if ASYNC_VIA_BUILTIN
  __builtin_amdgcn_s_wait_asynccnt(0);
#else
  asm volatile("s_wait_asynccnt 0x0" ::: "memory");
#endif
}

// ---------------------------------------------------------------------------
// f32 -> bf16 conversion (plain)
// ---------------------------------------------------------------------------
__global__ void cvt_f32_bf16(const float* __restrict__ in, bf_t* __restrict__ out,
                             size_t n) {
  size_t i = (size_t)blockIdx.x * blockDim.x + threadIdx.x;
  size_t stride = (size_t)gridDim.x * blockDim.x;
  for (; i < n; i += stride) out[i] = (bf_t)in[i];
}

// f32 (R x C, row-major) -> bf16 transposed (C x R): out[c][r] = in[r][c].
// Dims multiples of 32. Block 256 thr, grid (C/32, R/32).
__global__ __launch_bounds__(256)
void cvt_transpose_f32_bf16(const float* __restrict__ in, bf_t* __restrict__ out,
                            int R, int C) {
  __shared__ float t[32][33];
  const int bx = blockIdx.x * 32;   // column block (output row block)
  const int by = blockIdx.y * 32;   // row block
  const int tx = threadIdx.x & 31, ty = threadIdx.x >> 5;
#pragma unroll
  for (int i = ty; i < 32; i += 8) t[i][tx] = in[(size_t)(by + i) * C + bx + tx];
  __syncthreads();
#pragma unroll
  for (int i = ty; i < 32; i += 8)
    out[(size_t)(bx + i) * R + by + tx] = (bf_t)t[tx][i];
}

// ---------------------------------------------------------------------------
// WMMA fragment helpers (layouts per ISA 7.12.2).
// ---------------------------------------------------------------------------
__device__ __forceinline__ v16bf ld_frag(const bf_t* rowp, int koff, int hh) {
  v16bf f;
  const bf_t* p = rowp + koff + hh * 8;
#pragma unroll
  for (int e = 0; e < 8; ++e) { f[e] = p[e]; f[e + 8] = p[e + 16]; }
  return f;
}

__device__ __forceinline__ v8f wmma_bf16(v16bf a, v16bf b, v8f c) {
  return __builtin_amdgcn_wmma_f32_16x16x32_bf16(false, a, false, b, (short)0, c,
                                                 false, false);
}

// ---------------------------------------------------------------------------
// Tiled bf16 GEMM: C[M,N] = A[M,K] * Bt[N,K]^T  (Bt pre-transposed, K%32==0).
// Block 256 thr = 8 waves; tile 128x128x32, double-buffered async LDS staging.
// Out-of-range rows/cols are address-clamped: a clamped A row only feeds C rows
// that are never stored, likewise clamped Bt rows -> no zero-fill needed.
// ---------------------------------------------------------------------------
__global__ __launch_bounds__(256)
void gemm_bf16_wmma(const bf_t* __restrict__ A, const bf_t* __restrict__ Bt,
                    bf_t* __restrict__ Cb, float* __restrict__ Cf,
                    const float* __restrict__ bias,
                    int M, int N, int K, int out_bf16, int add_bias) {
  __shared__ __align__(16) bf_t As[2][128][40];   // [buf][m][k] pitch 40
  __shared__ __align__(16) bf_t Bs[2][128][40];   // [buf][n][k]

  const int tid  = threadIdx.x;
  const int lane = tid & 31;
  const int w    = tid >> 5;
  const int r    = lane & 15;
  const int hh   = lane >> 4;
  const int wm   = w & 3;
  const int wn   = w >> 2;
  const int bm   = blockIdx.x * 128;
  const int bn   = blockIdx.y * 128;

  v8f acc[2][4];
#pragma unroll
  for (int ms = 0; ms < 2; ++ms)
#pragma unroll
    for (int ns = 0; ns < 4; ++ns)
#pragma unroll
      for (int v = 0; v < 8; ++v) acc[ms][ns][v] = 0.f;

  // stage one 128x32 A tile + 128x32 B tile into buffer p (async).
  auto stage = [&](int k0, int p) {
#pragma unroll
    for (int u = 0; u < 2; ++u) {
      int c   = tid + u * 256;          // 0..511
      int row = c >> 2, seg = c & 3;    // seg = 16B chunk within 64B row
      int m = bm + row; if (m >= M) m = M - 1;
      async_copy16(A  + (size_t)m * K + k0 + seg * 8, &As[p][row][seg * 8]);
      int n = bn + row; if (n >= N) n = N - 1;
      async_copy16(Bt + (size_t)n * K + k0 + seg * 8, &Bs[p][row][seg * 8]);
    }
  };

  stage(0, 0);
  wait_async0();
  __syncthreads();

  int p = 0;
  for (int k0 = 0; k0 < K; k0 += 32) {
    if (k0 + 32 < K) stage(k0 + 32, p ^ 1);   // prefetch next tile (ASYNCcnt)

    v16bf af[2], bfr[4];
#pragma unroll
    for (int ms = 0; ms < 2; ++ms)
      af[ms] = ld_frag(&As[p][wm * 32 + ms * 16 + r][0], 0, hh);
#pragma unroll
    for (int ns = 0; ns < 4; ++ns)
      bfr[ns] = ld_frag(&Bs[p][wn * 64 + ns * 16 + r][0], 0, hh);
#pragma unroll
    for (int ms = 0; ms < 2; ++ms)
#pragma unroll
      for (int ns = 0; ns < 4; ++ns)
        acc[ms][ns] = wmma_bf16(af[ms], bfr[ns], acc[ms][ns]);

    wait_async0();
    __syncthreads();
    p ^= 1;
  }

#pragma unroll
  for (int ms = 0; ms < 2; ++ms)
#pragma unroll
    for (int ns = 0; ns < 4; ++ns)
#pragma unroll
      for (int v = 0; v < 8; ++v) {
        int m = bm + wm * 32 + ms * 16 + v + hh * 8;
        int n = bn + wn * 64 + ns * 16 + r;
        if (m < M && n < N) {
          float val = acc[ms][ns][v];
          if (out_bf16) Cb[(size_t)m * N + n] = (bf_t)val;
          else          Cf[(size_t)m * N + n] = add_bias ? val + bias[n] : val;
        }
      }
}

// ---------------------------------------------------------------------------
// Fused attention. Block = (b, h, 64 query rows); 128 threads = 4 waves, each
// wave owns 16 query rows. Text(+relpos)/image/audio streams, all on WMMA.
// ---------------------------------------------------------------------------
__global__ __launch_bounds__(128)
void attention_wmma(const bf_t* __restrict__ Q,     // (B*NQ, 1024)
                    const bf_t* __restrict__ Ktx,   // (B*125, 1024)
                    const bf_t* __restrict__ Vtx,
                    const bf_t* __restrict__ Kip,
                    const bf_t* __restrict__ Vip,
                    const bf_t* __restrict__ Kap,
                    const bf_t* __restrict__ Vap,
                    const float* __restrict__ relk,  // (33,64) f32
                    const float* __restrict__ relv,
                    const float* __restrict__ alpha_p,
                    const float* __restrict__ beta_p,
                    bf_t* __restrict__ Out) {        // (B*NQ, 1024)
  __shared__ __align__(16) bf_t Kt [80][72];    // text K  [j][d], rows 77..79 zero
  __shared__ __align__(16) bf_t Vts[64][104];   // text V^T [d][j], cols 77..95 zero
  __shared__ __align__(16) bf_t Ki [16][72];    // image K [j][d]
  __shared__ __align__(16) bf_t Vis[64][40];    // image V^T [d][j], cols 16..31 zero
  __shared__ __align__(16) bf_t Ka [32][72];    // audio K [j][d]
  __shared__ __align__(16) bf_t Vas[64][40];    // audio V^T [d][j]
  __shared__ __align__(16) bf_t RK [33][64];    // rel_k bf16
  __shared__ __align__(16) bf_t RV [33][64];    // rel_v bf16
  __shared__ __align__(16) bf_t Ps [4][16][104];// per-wave P tile (+qrel overlay)

  const int tid = threadIdx.x;
  const int qb  = blockIdx.x & 15;
  const int h   = (blockIdx.x >> 4) & 15;
  const int b   = blockIdx.x >> 8;

  // ---- zero the zero-padded staging arrays ----
  for (int i = tid; i < 80 * 72;  i += 128) (&Kt [0][0])[i] = (bf_t)0.f;
  for (int i = tid; i < 64 * 104; i += 128) (&Vts[0][0])[i] = (bf_t)0.f;
  for (int i = tid; i < 64 * 40;  i += 128) (&Vis[0][0])[i] = (bf_t)0.f;
  __syncthreads();

  const size_t cbase = (size_t)b * kCTX * kQD + (size_t)h * kHD;

  // ---- row-major K tiles via async global->LDS (16B chunks) ----
  for (int c = tid; c < kTEXT * 8; c += 128) {          // 77 rows x 8 chunks
    int j = c >> 3, seg = c & 7;
    async_copy16(Ktx + cbase + (size_t)j * kQD + seg * 8, &Kt[j][seg * 8]);
  }
  for (int c = tid; c < kIMG * 8; c += 128) {
    int j = c >> 3, seg = c & 7;
    async_copy16(Kip + cbase + (size_t)(kTEXT + j) * kQD + seg * 8,
                 &Ki[j][seg * 8]);
  }
  for (int c = tid; c < kAUD * 8; c += 128) {
    int j = c >> 3, seg = c & 7;
    async_copy16(Kap + cbase + (size_t)(kTEXT + kIMG + j) * kQD + seg * 8,
                 &Ka[j][seg * 8]);
  }

  // ---- transposed V tiles + bf16 rel tables (manual) ----
  for (int i = tid; i < kTEXT * kHD; i += 128) {
    int j = i >> 6, d = i & 63;
    Vts[d][j] = Vtx[cbase + (size_t)j * kQD + d];
  }
  for (int i = tid; i < kIMG * kHD; i += 128) {
    int j = i >> 6, d = i & 63;
    Vis[d][j] = Vip[cbase + (size_t)(kTEXT + j) * kQD + d];
  }
  for (int i = tid; i < kAUD * kHD; i += 128) {
    int j = i >> 6, d = i & 63;
    Vas[d][j] = Vap[cbase + (size_t)(kTEXT + kIMG + j) * kQD + d];
  }
  for (int i = tid; i < 33 * 64; i += 128) {
    int dd = i >> 6, d = i & 63;
    RK[dd][d] = (bf_t)relk[i];
    RV[dd][d] = (bf_t)relv[i];
  }
  wait_async0();
  __syncthreads();

  const int lane = tid & 31, w = tid >> 5;
  const int r = lane & 15, hh = lane >> 4;
  const int q0 = qb * 64 + w * 16;           // first query row of this wave

  // ---- Q fragments (D=64 -> two 32-wide k-steps), straight from global ----
  const bf_t* Qrow = Q + ((size_t)b * kNQ + q0 + r) * kQD + h * kHD;
  v16bf aq[2];
#pragma unroll
  for (int s = 0; s < 2; ++s) aq[s] = ld_frag(Qrow, s * 32, hh);

  // ---- qrel[i][dd] = dot(q_i, rel_k[dd])  (overlay on this wave's Ps) ----
  float* qrelw = (float*)(&Ps[w][0][0]);
  for (int dd = 0; dd < 33; ++dd) {
    float acc = 0.f;
#pragma unroll
    for (int s = 0; s < 2; ++s)
#pragma unroll
      for (int e = 0; e < 16; ++e) {
        int d = s * 32 + (e & 7) + ((e >> 3) << 4) + hh * 8;
        acc += (float)aq[s][e] * (float)RK[dd][d];
      }
    acc += __shfl_xor(acc, 16, 32);          // combine the two K-halves of row r
    if (hh == 0) qrelw[r * 33 + dd] = acc;
  }
  __syncthreads();

  // ================= TEXT stream (77 keys, rel-pos bias) =================
  float sv[5][8];
#pragma unroll
  for (int t = 0; t < 5; ++t) {
    v8f c = {0.f, 0.f, 0.f, 0.f, 0.f, 0.f, 0.f, 0.f};
#pragma unroll
    for (int s = 0; s < 2; ++s) {
      v16bf bk = ld_frag(&Kt[t * 16 + r][0], s * 32, hh);
      c = wmma_bf16(aq[s], bk, c);
    }
#pragma unroll
    for (int v = 0; v < 8; ++v) sv[t][v] = c[v];
  }
#pragma unroll
  for (int t = 0; t < 5; ++t) {
    int j = t * 16 + r;
#pragma unroll
    for (int v = 0; v < 8; ++v) {
      int i = v + hh * 8;
      if (j < kTEXT) {
        int qi = q0 + i;
        int dd = j - qi; dd = dd < -16 ? -16 : (dd > 16 ? 16 : dd); dd += 16;
        sv[t][v] = (sv[t][v] + qrelw[i * 33 + dd]) * kSCALE;
      } else sv[t][v] = -3.0e38f;
    }
  }
  float prob[5][8];
#pragma unroll
  for (int v = 0; v < 8; ++v) {
    float m = sv[0][v];
#pragma unroll
    for (int t = 1; t < 5; ++t) m = fmaxf(m, sv[t][v]);
#pragma unroll
    for (int msk = 1; msk < 16; msk <<= 1) m = fmaxf(m, __shfl_xor(m, msk, 32));
    float sum = 0.f;
#pragma unroll
    for (int t = 0; t < 5; ++t) {
      float e = (sv[t][v] < -1.0e38f) ? 0.f : __expf(sv[t][v] - m);
      prob[t][v] = e; sum += e;
    }
#pragma unroll
    for (int msk = 1; msk < 16; msk <<= 1) sum += __shfl_xor(sum, msk, 32);
    float inv = 1.f / sum;
#pragma unroll
    for (int t = 0; t < 5; ++t) prob[t][v] *= inv;
  }
  __syncthreads();                            // done with qrel overlay
  for (int idx = lane; idx < 16 * 104; idx += 32) (&Ps[w][0][0])[idx] = (bf_t)0.f;
#pragma unroll
  for (int t = 0; t < 5; ++t) {
    int j = t * 16 + r;
    if (j < kTEXT)
#pragma unroll
      for (int v = 0; v < 8; ++v) Ps[w][v + hh * 8][j] = (bf_t)prob[t][v];
  }

  // O += P @ V  (16x96 zero-padded x 96x64)
  v8f oacc[4];
#pragma unroll
  for (int n = 0; n < 4; ++n)
#pragma unroll
    for (int v = 0; v < 8; ++v) oacc[n][v] = 0.f;

  v16bf ap[3];
#pragma unroll
  for (int ks = 0; ks < 3; ++ks) ap[ks] = ld_frag(&Ps[w][r][0], ks * 32, hh);
#pragma unroll
  for (int n = 0; n < 4; ++n)
#pragma unroll
    for (int ks = 0; ks < 3; ++ks) {
      v16bf bv = ld_frag(&Vts[n * 16 + r][0], ks * 32, hh);
      oacc[n] = wmma_bf16(ap[ks], bv, oacc[n]);
    }
  // O += P @ rel_v[dist]
#pragma unroll
  for (int n = 0; n < 4; ++n)
#pragma unroll
    for (int v = 0; v < 8; ++v) {
      int i = v + hh * 8, d = n * 16 + r, qi = q0 + i;
      float a2 = 0.f;
      for (int j = 0; j < kTEXT; ++j) {
        int dd = j - qi; dd = dd < -16 ? -16 : (dd > 16 ? 16 : dd); dd += 16;
        a2 += (float)Ps[w][i][j] * (float)RV[dd][d];
      }
      oacc[n][v] += a2;
    }

  // ================= IMAGE stream (16 keys, gated) =================
  {
    float ga = tanhf(alpha_p[0]) + 1.f;
    v8f c = {0.f, 0.f, 0.f, 0.f, 0.f, 0.f, 0.f, 0.f};
#pragma unroll
    for (int s = 0; s < 2; ++s) {
      v16bf bk = ld_frag(&Ki[r][0], s * 32, hh);
      c = wmma_bf16(aq[s], bk, c);
    }
    float pv[8];
#pragma unroll
    for (int v = 0; v < 8; ++v) {
      float sc = c[v] * kSCALE;
      float m = sc;
#pragma unroll
      for (int msk = 1; msk < 16; msk <<= 1) m = fmaxf(m, __shfl_xor(m, msk, 32));
      float e = __expf(sc - m);
      float sum = e;
#pragma unroll
      for (int msk = 1; msk < 16; msk <<= 1) sum += __shfl_xor(sum, msk, 32);
      pv[v] = e / sum * ga;
    }
    for (int idx = lane; idx < 16 * 32; idx += 32)
      Ps[w][idx >> 5][idx & 31] = (bf_t)0.f;
#pragma unroll
    for (int v = 0; v < 8; ++v) Ps[w][v + hh * 8][r] = (bf_t)pv[v];
    v16bf api = ld_frag(&Ps[w][r][0], 0, hh);
#pragma unroll
    for (int n = 0; n < 4; ++n) {
      v16bf bv = ld_frag(&Vis[n * 16 + r][0], 0, hh);
      oacc[n] = wmma_bf16(api, bv, oacc[n]);
    }
  }

  // ================= AUDIO stream (32 keys, gated) =================
  {
    float gb = tanhf(beta_p[0]) + 1.f;
    float sa[2][8];
#pragma unroll
    for (int t = 0; t < 2; ++t) {
      v8f c = {0.f, 0.f, 0.f, 0.f, 0.f, 0.f, 0.f, 0.f};
#pragma unroll
      for (int s = 0; s < 2; ++s) {
        v16bf bk = ld_frag(&Ka[t * 16 + r][0], s * 32, hh);
        c = wmma_bf16(aq[s], bk, c);
      }
#pragma unroll
      for (int v = 0; v < 8; ++v) sa[t][v] = c[v] * kSCALE;
    }
    float pa[2][8];
#pragma unroll
    for (int v = 0; v < 8; ++v) {
      float m = fmaxf(sa[0][v], sa[1][v]);
#pragma unroll
      for (int msk = 1; msk < 16; msk <<= 1) m = fmaxf(m, __shfl_xor(m, msk, 32));
      float e0 = __expf(sa[0][v] - m), e1 = __expf(sa[1][v] - m);
      float sum = e0 + e1;
#pragma unroll
      for (int msk = 1; msk < 16; msk <<= 1) sum += __shfl_xor(sum, msk, 32);
      float inv = gb / sum;
      pa[0][v] = e0 * inv; pa[1][v] = e1 * inv;
    }
#pragma unroll
    for (int t = 0; t < 2; ++t)
#pragma unroll
      for (int v = 0; v < 8; ++v) Ps[w][v + hh * 8][t * 16 + r] = (bf_t)pa[t][v];
    v16bf apa = ld_frag(&Ps[w][r][0], 0, hh);
#pragma unroll
    for (int n = 0; n < 4; ++n) {
      v16bf bv = ld_frag(&Vas[n * 16 + r][0], 0, hh);
      oacc[n] = wmma_bf16(apa, bv, oacc[n]);
    }
  }

  // ---- write O tile (bf16, consumed by final GEMM) ----
#pragma unroll
  for (int n = 0; n < 4; ++n)
#pragma unroll
    for (int v = 0; v < 8; ++v) {
      size_t row = (size_t)b * kNQ + q0 + v + hh * 8;
      int    col = h * kHD + n * 16 + r;
      Out[row * kQD + col] = (bf_t)oacc[n][v];
    }
}

// ---------------------------------------------------------------------------
// Launcher: cvt(+weight transpose) -> Q proj -> 6 KV proj -> attention -> out.
// Workspace ~276 MB of d_ws.
// ---------------------------------------------------------------------------
extern "C" void kernel_launch(void* const* d_in, const int* in_sizes, int n_in,
                              void* d_out, int out_size, void* d_ws, size_t ws_size,
                              hipStream_t stream) {
  (void)in_sizes; (void)n_in; (void)out_size; (void)ws_size;
  const float* x     = (const float*)d_in[0];
  const float* ctx   = (const float*)d_in[1];
  const float* Wsrc[8] = {(const float*)d_in[2], (const float*)d_in[3],
                          (const float*)d_in[4], (const float*)d_in[5],
                          (const float*)d_in[6], (const float*)d_in[7],
                          (const float*)d_in[8], (const float*)d_in[9]};
  const float* bo    = (const float*)d_in[10];
  const float* relk  = (const float*)d_in[11];
  const float* relv  = (const float*)d_in[12];
  const float* alpha = (const float*)d_in[13];
  const float* beta  = (const float*)d_in[14];
  float* out = (float*)d_out;

  char* ws = (char*)d_ws;
  size_t off = 0;
  auto take = [&](size_t elems) -> bf_t* {
    bf_t* p = (bf_t*)(ws + off);
    off += ((elems * sizeof(bf_t)) + 255) & ~(size_t)255;
    return p;
  };

  const size_t nX = (size_t)kB * kNQ * kQD;        // 33.5M
  const size_t nC = (size_t)kB * kCTX * kQD;       // 4.1M
  bf_t* xb = take(nX);
  bf_t* cb = take(nC);
  bf_t* wb[8];                                     // transposed: [n][k]
  for (int i = 0; i < 8; ++i) wb[i] = take((size_t)kQD * kQD);
  bf_t* Qb = take(nX);
  bf_t* KVb[6];
  for (int i = 0; i < 6; ++i) KVb[i] = take(nC);
  bf_t* outb = take(nX);

  // conversions (+ one-time weight transposes so GEMM B tiles are contiguous)
  cvt_f32_bf16<<<8192, 256, 0, stream>>>(x, xb, nX);
  cvt_f32_bf16<<<4096, 256, 0, stream>>>(ctx, cb, nC);
  dim3 gt(kQD / 32, kQD / 32);
  for (int i = 0; i < 8; ++i)
    cvt_transpose_f32_bf16<<<gt, 256, 0, stream>>>(Wsrc[i], wb[i], kQD, kQD);

  dim3 blk(256);
  const int Mq = kB * kNQ;      // 32768
  const int Mc = kB * kCTX;     // 4000
  dim3 gq((Mq + 127) / 128, kQD / 128);
  dim3 gc((Mc + 127) / 128, kQD / 128);

  // Q = x @ Wq   (bf16 out)
  gemm_bf16_wmma<<<gq, blk, 0, stream>>>(xb, wb[0], Qb, nullptr, nullptr,
                                         Mq, kQD, kQD, 1, 0);
  // 6 context projections over the full 125 rows (streams slice later)
  for (int s = 0; s < 6; ++s)
    gemm_bf16_wmma<<<gc, blk, 0, stream>>>(cb, wb[1 + s], KVb[s], nullptr, nullptr,
                                           Mc, kQD, kQD, 1, 0);

  // fused attention: (b, h, 64-query block)
  attention_wmma<<<kB * kNH * (kNQ / 64), 128, 0, stream>>>(
      Qb, KVb[0], KVb[1], KVb[2], KVb[3], KVb[4], KVb[5],
      relk, relv, alpha, beta, outb);

  // out = attn_out @ Wo + bo  (f32 out)
  gemm_bf16_wmma<<<gq, blk, 0, stream>>>(outb, wb[7], nullptr, out, bo,
                                         Mq, kQD, kQD, 0, 1);
}
